// Main_Module_41248865911490
// MI455X (gfx1250) — compile-verified, matrix operand order checked by reading, not run on hardware
//
#include <hip/hip_runtime.h>
#include <cmath>

// ---------------------------------------------------------------------------
// MI455X (gfx1250, wave32).
//   loss = 0.5*mean( (NUDFT(pred)-NUDFT(gt))^2 ) == 0.5*mean( NUDFT(pred-gt)^2 )
// Stage 1: SIREN eval -> diff volume (f16), WMMA f16 GEMMs, TDM weight staging.
// Stage 2: NUDFT of diff volume (WMMA, K split in 4 slabs, no atomics).
// Stage 3: deterministic square-reduce -> scalar loss.
// ---------------------------------------------------------------------------

typedef __attribute__((ext_vector_type(16))) _Float16     v16h;
typedef __attribute__((ext_vector_type(8)))  float        v8f;
typedef __attribute__((ext_vector_type(4)))  unsigned int v4u;
typedef __attribute__((ext_vector_type(8)))  int          v8i;
typedef __attribute__((ext_vector_type(4)))  int          v4i;

constexpr int NX = 128, NY = 128, NZ = 64;
constexpr int NPTS = NX * NY * NZ;          // 1,048,576
constexpr int DH   = 128;                   // hidden width
constexpr int EMB  = 64;                    // fourier features (x2 for sin/cos)
constexpr int KTOT = 2048;                  // 8 spokes * 256 samples
constexpr int PTOT = NX * NY;               // 16384 pixels per slice
constexpr int NSLAB = 4;                    // pixel-dim slabs for NUDFT parallelism
constexpr int PSLAB = PTOT / NSLAB;         // 4096
constexpr int KRKI  = 2 * NZ * KTOT;        // 262144 partial k-space elems per slab

constexpr int AS = 136;   // LDS activation row stride (halves): 272B, 16B aligned, 4-bank skew
constexpr int WS = 136;   // LDS weight row stride (halves) — matches TDM pad below
constexpr int PS = 136;   // LDS pixel-dim stride (halves) for NUDFT tiles

#define HAVE_TDM __has_builtin(__builtin_amdgcn_tensor_load_to_lds)

// -------------------- weight f32 -> f16 conversion --------------------------
__global__ void prep_weights_kernel(const float* __restrict__ W0,
                                    const float* __restrict__ W1,
                                    const float* __restrict__ W2,
                                    const float* __restrict__ W3,
                                    _Float16* __restrict__ w16) {
    int idx = blockIdx.x * 256 + threadIdx.x;          // 4*128*128 = 65536 total
    if (idx >= 4 * DH * DH) return;
    const float* Ws[4] = {W0, W1, W2, W3};
    w16[idx] = (_Float16)Ws[idx >> 14][idx & 16383];
}

// -------------------- SIREN eval + diff volume (WMMA) ------------------------
// 128 threads = 4 waves; block owns 128 grid points; each wave owns TWO 16-row
// M-tiles so every B-fragment (weights) feeds two WMMAs.
__global__ __launch_bounds__(128)
void siren_diff_kernel(const float* __restrict__ Bmat,     // [64,3]
                       const _Float16* __restrict__ w16,   // [4][128][128] (row = out neuron)
                       const float* __restrict__ b0,
                       const float* __restrict__ b1,
                       const float* __restrict__ b2,
                       const float* __restrict__ b3,
                       const float* __restrict__ W4,       // [1,128]
                       const float* __restrict__ b4,
                       const float* __restrict__ image,    // [NPTS] x-major
                       _Float16* __restrict__ diff16) {    // [NPTS]
    __shared__ _Float16 actA[128 * AS];
    __shared__ _Float16 actB[128 * AS];
    __shared__ _Float16 wlds[DH * WS];
    __shared__ float    blds[DH];

    const int tid  = threadIdx.x;
    const int lane = tid & 31;
    const int wave = tid >> 5;                 // 0..3
    const long long n0 = (long long)blockIdx.x * 128;

    // ---- Fourier-feature encode: feat = [sin(p) | cos(p)], p = 2*pi*x . B[j]
    for (int idx = tid; idx < 128 * EMB; idx += 128) {
        const int pt = idx >> 6;
        const int j  = idx & 63;
        const long long n = n0 + pt;
        const int x = (int)(n >> 13);          // n / (NY*NZ)
        const int y = (int)((n >> 6) & 127);
        const int z = (int)(n & 63);
        const float gx = ((float)x + 0.5f) * (1.0f / 128.0f);
        const float gy = ((float)y + 0.5f) * (1.0f / 128.0f);
        const float gz = ((float)z + 0.5f) * (1.0f / 64.0f);
        const float p = 6.28318530717958647692f *
            (gx * Bmat[j * 3 + 0] + gy * Bmat[j * 3 + 1] + gz * Bmat[j * 3 + 2]);
        float s, c;
        __sincosf(p, &s, &c);
        actA[pt * AS + j]       = (_Float16)s;
        actA[pt * AS + EMB + j] = (_Float16)c;
    }

    const float* biases[4] = {b0, b1, b2, b3};

    for (int layer = 0; layer < 4; ++layer) {
        // ---- stage layer weights (128x128 f16) into LDS ----
#if HAVE_TDM
        // Tensor Data Mover: 2D tile, 2-byte elements, tensor_dim0 = tile_dim0
        // = 128, stride 128; pad_enable inserts 4 DWORDs (8 halves) after every
        // 64 DWORDs (one row) -> padded LDS row stride of 136 halves (== WS).
        if (wave == 0) {
            const unsigned long long ga =
                (unsigned long long)(uintptr_t)(w16 + layer * (DH * DH));
            const unsigned ldsa = (unsigned)(uintptr_t)(&wlds[0]);
            // group0: count=1 | lds_addr | global_addr[56:0] | type=2 (bits 127:126)
            v4u g0 = {1u, ldsa, (unsigned)ga,
                      ((unsigned)((ga >> 32) & 0x01FFFFFFull)) | 0x80000000u};
            // group1: data_size=1(2B) | pad_en | pad_interval=5(64 dw) | pad_amount=3(4 dw)
            //         tensor_dim0=128, tensor_dim1=128, tile_dim0=128, tile_dim1=128,
            //         tensor_dim0_stride=128
            v8i g1 = {(int)0x07510000,
                      (int)(128u << 16),   // tensor_dim0 lo16 in [63:48]
                      (int)(128u << 16),   // dim0 hi=0, tensor_dim1 lo16 in [95:80]
                      (int)(128u << 16),   // dim1 hi=0, tile_dim0 in [127:112]
                      128,                 // tile_dim1 in [143:128], tile_dim2=0
                      128,                 // tensor_dim0_stride lo32
                      0, 0};
            v4i g2 = {0, 0, 0, 0};
            v4i g3 = {0, 0, 0, 0};
            v8i gz = {0, 0, 0, 0, 0, 0, 0, 0};
            __builtin_amdgcn_tensor_load_to_lds(g0, g1, g2, g3, gz, 0);
            __builtin_amdgcn_s_wait_tensorcnt(0);
        }
#else
        {
            const _Float16* wsrc = w16 + layer * (DH * DH);
            for (int idx = tid; idx < (DH * DH) / 8; idx += 128) {
                const int nrow = (idx * 8) >> 7;
                const int kcol = (idx * 8) & 127;
                *(uint4*)(&wlds[nrow * WS + kcol]) =
                    *(const uint4*)(wsrc + nrow * DH + kcol);
            }
        }
#endif
        if (tid < DH) blds[tid] = biases[layer][tid];
        __syncthreads();   // weights + (first iter) encode visible

        const _Float16* src = (layer & 1) ? actB : actA;
        _Float16*       dst = (layer & 1) ? actA : actB;

        const int m0  = wave * 32;           // two M-tiles per wave
        const int r   = lane & 15;
        const int kb8 = (lane >> 4) * 8;     // wave32 A/B lane k-offset

        v8f acc0[8] = {};
        v8f acc1[8] = {};
#pragma unroll
        for (int ks = 0; ks < 4; ++ks) {
            const int kbase = ks * 32 + kb8;
            v16h a0, a1;
#pragma unroll
            for (int e = 0; e < 16; ++e) {
                const int kk = kbase + (e < 8 ? e : e + 8);
                a0[e] = src[(m0 + r) * AS + kk];
                a1[e] = src[(m0 + 16 + r) * AS + kk];
            }
#pragma unroll
            for (int t = 0; t < 8; ++t) {
                const int ncol = t * 16 + r;
                v16h b;
#pragma unroll
                for (int e = 0; e < 16; ++e)
                    b[e] = wlds[ncol * WS + kbase + (e < 8 ? e : e + 8)];
                acc0[t] = __builtin_amdgcn_wmma_f32_16x16x32_f16(
                    false, a0, false, b, (short)0, acc0[t], false, false);
                acc1[t] = __builtin_amdgcn_wmma_f32_16x16x32_f16(
                    false, a1, false, b, (short)0, acc1[t], false, false);
            }
        }

        // D layout: VGPR v -> row (v + 8*(lane>=16)), lane&15 -> column
#pragma unroll
        for (int t = 0; t < 8; ++t) {
            const int ncol = t * 16 + r;
            const float bias = blds[ncol];
#pragma unroll
            for (int v = 0; v < 8; ++v) {
                const int m = v + ((lane >> 4) << 3);
                dst[(m0 + m) * AS + ncol] =
                    (_Float16)__sinf(30.0f * (acc0[t][v] + bias));
                dst[(m0 + 16 + m) * AS + ncol] =
                    (_Float16)__sinf(30.0f * (acc1[t][v] + bias));
            }
        }
        __syncthreads();   // all wlds/src reads + dst writes complete
    }

    // final 128->1 layer (layer3 wrote actA), then diff vs ground truth
    {
        float sum = b4[0];
#pragma unroll 8
        for (int k = 0; k < DH; ++k)
            sum += (float)actA[tid * AS + k] * W4[k];
        const long long n = n0 + tid;
        diff16[n] = (_Float16)(sum - image[n]);
    }
}

// -------------------- NUDFT partials of diff volume (WMMA) -------------------
// blockIdx.x = ktile(128) x slab(4) -> 512 blocks. Block owns 16 k-samples and
// one 4096-pixel slab. 8 waves = {4 z-tiles} x {cos, sin}. Each wave writes its
// 16x16 f32 partial sums to krki[slab] exactly once (no atomics).
__global__ __launch_bounds__(256)
void nudft_partial_kernel(const _Float16* __restrict__ diff16,  // [NPTS]
                          const float* __restrict__ ktraj,      // [2, 2048]
                          float* __restrict__ krki) {           // [4][2][64][2048]
    __shared__ _Float16 trigC[16 * PS];
    __shared__ _Float16 trigS[16 * PS];
    __shared__ _Float16 aLds[64 * PS];
    __shared__ float    kxs[16], kys[16];

    const int tid  = threadIdx.x;
    const int lane = tid & 31;
    const int wave = tid >> 5;
    const int k0   = (blockIdx.x & 127) * 16;
    const int slab = blockIdx.x >> 7;          // 0..3
    const int pbeg = slab * PSLAB;

    if (tid < 16) {
        kxs[tid] = ktraj[k0 + tid];
        kys[tid] = ktraj[KTOT + k0 + tid];
    }

    const int mtile  = wave >> 1;              // z-tile 0..3
    const int useSin = wave & 1;               // 0 = real(cos), 1 = imag(sin)
    const int zrow   = mtile * 16 + (lane & 15);
    const int kb8    = (lane >> 4) * 8;
    const int ncol   = lane & 15;

    v8f acc = {};

    for (int p0 = pbeg; p0 < pbeg + PSLAB; p0 += 128) {
        __syncthreads();   // previous chunk's LDS reads done (also covers kxs)

        // phase tiles: 128 pixels x 16 k-samples (8 sincos per thread)
        for (int idx = tid; idx < 2048; idx += 256) {
            const int pl = idx >> 4;
            const int kl = idx & 15;
            const int pg = p0 + pl;
            const float x = (float)(pg >> 7);
            const float y = (float)(pg & 127);
            float s, c;
            __sincosf(kxs[kl] * x + kys[kl] * y, &s, &c);
            trigC[kl * PS + pl] = (_Float16)c;
            trigS[kl * PS + pl] = (_Float16)s;
        }
        // slice tile: 64 z x 128 pixels (z fastest -> coalesced global reads)
        for (int idx = tid; idx < 8192; idx += 256) {
            const int z  = idx & 63;
            const int pl = idx >> 6;
            const int pg = p0 + pl;
            const int x = pg >> 7, y = pg & 127;
            aLds[z * PS + pl] = diff16[(size_t)x * 8192 + (size_t)y * 64 + z];
        }
        __syncthreads();

        const _Float16* trig = useSin ? trigS : trigC;
#pragma unroll
        for (int ks = 0; ks < 4; ++ks) {
            const int pb = ks * 32 + kb8;
            v16h a, b;
#pragma unroll
            for (int e = 0; e < 16; ++e) {
                const int pp = pb + (e < 8 ? e : e + 8);
                a[e] = aLds[zrow * PS + pp];
                b[e] = trig[ncol * PS + pp];
            }
            acc = __builtin_amdgcn_wmma_f32_16x16x32_f16(
                false, a, false, b, (short)0, acc, false, false);
        }
    }

    // write 16x16 partial tile (sign of imag part irrelevant: it gets squared)
    float* dst = krki + ((size_t)slab * 2 + useSin) * (NZ * KTOT);
#pragma unroll
    for (int v = 0; v < 8; ++v) {
        const int z = mtile * 16 + v + ((lane >> 4) << 3);
        dst[z * KTOT + k0 + ncol] = acc[v];
    }
}

// -------------------- deterministic square-reduce ----------------------------
__global__ __launch_bounds__(256)
void square_reduce_kernel(const float* __restrict__ krki,   // [4][KRKI]
                          float* __restrict__ partials) {   // [256]
    __shared__ float red[256];
    const int tid = threadIdx.x;
    float local = 0.0f;
    for (int j = blockIdx.x * 256 + tid; j < KRKI; j += 256 * 256) {
        const float v = krki[j] + krki[KRKI + j] + krki[2 * KRKI + j] + krki[3 * KRKI + j];
        local += v * v;
    }
    red[tid] = local;
    __syncthreads();
    for (int s = 128; s > 0; s >>= 1) {
        if (tid < s) red[tid] += red[tid + s];
        __syncthreads();
    }
    if (tid == 0) partials[blockIdx.x] = red[0];
}

__global__ __launch_bounds__(256)
void final_loss_kernel(const float* __restrict__ partials, float* __restrict__ out) {
    __shared__ float red[256];
    const int tid = threadIdx.x;
    red[tid] = partials[tid];
    __syncthreads();
    for (int s = 128; s > 0; s >>= 1) {
        if (tid < s) red[tid] += red[tid + s];
        __syncthreads();
    }
    // loss = 0.5 * mean over [64,2048] of (dkr^2 + dki^2)
    if (tid == 0) out[0] = red[0] * (0.5f / 131072.0f);
}

// ----------------------------------------------------------------------------
extern "C" void kernel_launch(void* const* d_in, const int* in_sizes, int n_in,
                              void* d_out, int out_size, void* d_ws, size_t ws_size,
                              hipStream_t stream) {
    (void)in_sizes; (void)n_in; (void)out_size; (void)ws_size;

    // setup_inputs order:
    // 0:grid 1:B 2:W0 3:b0 4:W1 5:b1 6:W2 7:b2 8:W3 9:b3 10:W4 11:b4 12:image 13:ktraj
    const float* Bmat  = (const float*)d_in[1];
    const float* W0    = (const float*)d_in[2];
    const float* b0    = (const float*)d_in[3];
    const float* W1    = (const float*)d_in[4];
    const float* b1    = (const float*)d_in[5];
    const float* W2    = (const float*)d_in[6];
    const float* b2    = (const float*)d_in[7];
    const float* W3    = (const float*)d_in[8];
    const float* b3    = (const float*)d_in[9];
    const float* W4    = (const float*)d_in[10];
    const float* b4    = (const float*)d_in[11];
    const float* image = (const float*)d_in[12];
    const float* ktraj = (const float*)d_in[13];

    char* ws = (char*)d_ws;
    _Float16* w16      = (_Float16*)(ws);                       // 128 KiB
    _Float16* diff16   = (_Float16*)(ws + 131072);              // 2 MiB
    float*    krki     = (float*)   (ws + 131072 + 2097152);    // 4 MiB
    float*    partials = (float*)   (ws + 131072 + 2097152 + 4194304); // 1 KiB
    float*    loss     = (float*)d_out;

    prep_weights_kernel<<<256, 256, 0, stream>>>(W0, W1, W2, W3, w16);
    siren_diff_kernel<<<NPTS / 128, 128, 0, stream>>>(
        Bmat, w16, b0, b1, b2, b3, W4, b4, image, diff16);
    nudft_partial_kernel<<<(KTOT / 16) * NSLAB, 256, 0, stream>>>(diff16, ktraj, krki);
    square_reduce_kernel<<<256, 256, 0, stream>>>(krki, partials);
    final_loss_kernel<<<1, 256, 0, stream>>>(partials, loss);
}
